// NAFA_39771397161408
// MI455X (gfx1250) — matrix operand order repaired and checked by reading, not compiled, not run on hardware
//
#include <hip/hip_runtime.h>
#include <math.h>

typedef __attribute__((ext_vector_type(16))) _Float16 v16h;
typedef __attribute__((ext_vector_type(8)))  float    v8f;

#define B_ 8
#define T_ 2048
#define F_ 128
#define W_ 10
#define NW_ (T_ - W_ + 1)
#define EPS_ 1e-12f
#define RMAX_ (1.0f - 0.0001f)
#define RMIN_ 0.0001f

// ---------------- conv(F->1, k=5, same) + sigmoid ----------------
__global__ void k_conv_sigmoid(const float* __restrict__ x,
                               const float* __restrict__ w,
                               const float* __restrict__ bias,
                               float* __restrict__ score_raw) {
  __shared__ float wl[F_ * 5];
  int tid = threadIdx.x;
  for (int i = tid; i < F_ * 5; i += 256) wl[i] = w[i];
  __syncthreads();
  int g = blockIdx.x * 256 + tid;
  int b = g >> 11, t = g & (T_ - 1);
  float acc = bias[0];
  for (int k = 0; k < 5; ++k) {
    int tk = t + k - 2;
    if (tk < 0 || tk >= T_) continue;
    const float* xr = x + ((size_t)(b * T_ + tk) << 7);
    #pragma unroll 8
    for (int f = 0; f < F_; ++f) acc = fmaf(xr[f], wl[f * 5 + k], acc);
  }
  score_raw[g] = 1.0f / (1.0f + __expf(-acc));
}

// ---------------- per-batch sum & max of raw score ----------------
__global__ void k_batch_reduce(const float* __restrict__ raw,
                               float* __restrict__ sum0,
                               float* __restrict__ max0) {
  __shared__ float rs[256], rm[256];
  int b = blockIdx.x, tid = threadIdx.x;
  float s = 0.f, m = -1e30f;
  for (int t = tid; t < T_; t += 256) {
    float v = raw[b * T_ + t];
    s += v;
    m = fmaxf(m, v);
  }
  rs[tid] = s; rm[tid] = m;
  __syncthreads();
  for (int o = 128; o > 0; o >>= 1) {
    if (tid < o) { rs[tid] += rs[tid + o]; rm[tid] = fmaxf(rm[tid], rm[tid + o]); }
    __syncthreads();
  }
  if (tid == 0) { sum0[b] = rs[0]; max0[b] = rm[0]; }
}

// ---------------- score_norm (per-batch block) ----------------
__global__ void k_score_norm(const float* __restrict__ raw,
                             const float* __restrict__ sum0,
                             const float* __restrict__ max0,
                             float* __restrict__ score_out) {
  __shared__ float r1[256], r2[256];
  int b = blockIdx.x, tid = threadIdx.x;
  bool any_need = false;
  for (int bb = 0; bb < B_; ++bb)
    any_need = any_need || ((max0[bb] * (float)T_ / sum0[bb]) >= 1.0f);
  float scale = (float)T_ / sum0[b];
  float mx = max0[b] * scale;
  float inv_mx = (mx >= 1.0f) ? (1.0f / mx) : 1.0f;
  float s2 = 0.f, den = 0.f;
  float sc[8], iv[8];
  #pragma unroll
  for (int i = 0; i < 8; ++i) {
    int t = tid + i * 256;
    float v = raw[b * T_ + t] * scale * inv_mx;
    float msk = (v < RMAX_ && v > RMIN_) ? 1.0f : 0.0f;
    float in = (1.0f - v) * msk;
    sc[i] = v; iv[i] = in;
    s2 += v; den += in;
  }
  r1[tid] = s2; r2[tid] = den;
  __syncthreads();
  for (int o = 128; o > 0; o >>= 1) {
    if (tid < o) { r1[tid] += r1[tid + o]; r2[tid] += r2[tid + o]; }
    __syncthreads();
  }
  float DEN = r2[0];
  float dist = (float)T_ - r1[0];
  float alpha = (DEN > 0.f) ? fminf(dist / DEN, 1.0f) : 0.0f;
  float aa = alpha * ((any_need && dist >= 1.0f) ? 1.0f : 0.0f);
  #pragma unroll
  for (int i = 0; i < 8; ++i) {
    int t = tid + i * 256;
    score_out[b * T_ + t] = sc[i] + iv[i] * aa;
  }
}

// ------- feature = exp(x), transposed to f16 [b][n][k], tiled via LDS -------
__global__ void k_featT(const float* __restrict__ x, _Float16* __restrict__ featT) {
  __shared__ float tile[32][33];
  int b = blockIdx.z;
  int k0 = blockIdx.x << 5;   // T/32 tiles
  int f0 = blockIdx.y << 5;   // F/32 tiles
  int tx = threadIdx.x & 31, ty = threadIdx.x >> 5;   // 32 x 8
  #pragma unroll
  for (int i = 0; i < 4; ++i) {
    int r = ty + (i << 3);
    tile[r][tx] = x[((size_t)(b * T_ + k0 + r) << 7) + f0 + tx];   // coalesced in f
  }
  __syncthreads();
  #pragma unroll
  for (int i = 0; i < 4; ++i) {
    int r = ty + (i << 3);
    featT[((size_t)(b * F_ + f0 + r) << 11) + k0 + tx] =
        (_Float16)__expf(tile[tx][r]);                              // coalesced in k
  }
}

// ---------------- softmax row stats: max & sumexp of proj*score ----------------
__global__ void k_rowstats(const float* __restrict__ proj,
                           const float* __restrict__ score,
                           float* __restrict__ rowmax,
                           float* __restrict__ rowsum) {
  __shared__ float vals[T_];
  __shared__ float red[256];
  int i = blockIdx.x, b = blockIdx.y, tid = threadIdx.x;
  const float* pr = proj + ((size_t)(b * T_) + i) * T_;
  const float* sc = score + b * T_;
  float m = -1e30f;
  for (int j = tid; j < T_; j += 256) {
    float v = pr[j] * sc[j];
    vals[j] = v;
    m = fmaxf(m, v);
  }
  red[tid] = m;
  __syncthreads();
  for (int o = 128; o > 0; o >>= 1) {
    if (tid < o) red[tid] = fmaxf(red[tid], red[tid + o]);
    __syncthreads();
  }
  m = red[0];
  __syncthreads();
  float s = 0.f;
  for (int j = tid; j < T_; j += 256) s += __expf(vals[j] - m);
  red[tid] = s;
  __syncthreads();
  for (int o = 128; o > 0; o >>= 1) {
    if (tid < o) red[tid] += red[tid + o];
    __syncthreads();
  }
  if (tid == 0) { rowmax[b * T_ + i] = m; rowsum[b * T_ + i] = red[0]; }
}

// ---------------- fused softmax-weight build + WMMA GEMM ----------------
// 256 threads = 8 waves, 16 rows x 128 cols per WG. K-step = 128 (4 WMMAs/step),
// double-buffered 16x128 f16 A tile in LDS -> one barrier per step; next tile's
// proj loads + exps pipelined against current tile's WMMAs.
__global__ void k_gemm(const float* __restrict__ proj,
                       const _Float16* __restrict__ featT,
                       const float* __restrict__ score,
                       const float* __restrict__ rowmax,
                       const float* __restrict__ rowsum,
                       float* __restrict__ alignedWS,
                       float* __restrict__ outFeat) {
  __shared__ float sc[T_];
  __shared__ float rmax[16], rsinv[16];
  __shared__ __align__(16) _Float16 Abuf[2][16 * 128];
  int tid = threadIdx.x;
  int b = blockIdx.y, i0 = blockIdx.x << 4;
  for (int j = tid; j < T_; j += 256) sc[j] = score[b * T_ + j];
  if (tid < 16) {
    rmax[tid]  = rowmax[b * T_ + i0 + tid];
    rsinv[tid] = 1.0f / rowsum[b * T_ + i0 + tid];
  }
  __syncthreads();

  int lane = tid & 31, wave = tid >> 5;
  // A-builder role: 16 threads per row, 8 consecutive k each
  int am = tid >> 4, ak = (tid & 15) << 3;
  const float* pr = proj + ((size_t)(b * T_) + i0 + am) * T_ + ak;
  float rmA = rmax[am];
  // A-fragment role (contiguous k order, same as B): row = lane&15;
  // lane<16 -> K 0-15 ; lane>=16 -> K 16-31 (per 32-K slice)
  int arow = lane & 15;
  int koff = (lane >> 4) << 4;          // 0 or 16
  // B-fragment role: N = wave*16 + lane&15 ; half-wave = 16 consecutive K
  int n = (wave << 4) + (lane & 15);
  int kbh = (lane >> 4);                // +16 halves for upper half-wave
  const v16h* bbase = (const v16h*)(featT + ((size_t)(b * F_ + n) << 11));

  auto buildA = [&](int k0, _Float16* buf) {
    __builtin_prefetch(pr + k0 + 128, 0, 0);
    union { float4 v[2]; float f[8]; } pl;
    pl.v[0] = *(const float4*)(pr + k0);
    pl.v[1] = *(const float4*)(pr + k0 + 4);
    union { _Float16 h[8]; uint4 q; } pk;
    #pragma unroll
    for (int i = 0; i < 8; ++i)
      pk.h[i] = (_Float16)__expf(pl.f[i] * sc[k0 + ak + i] - rmA);
    *(uint4*)&buf[(am << 7) + ak] = pk.q;
  };

  v8f c = {};
  const int NSTEP = T_ / 128;           // 16
  buildA(0, Abuf[0]);
  for (int s = 0; s < NSTEP; ++s) {
    __syncthreads();                    // Abuf[p] ready; Abuf[p^1] fully consumed
    int p = s & 1;
    if (s + 1 < NSTEP) buildA((s + 1) << 7, Abuf[p ^ 1]);
    const _Float16* Ab = Abuf[p];
    #pragma unroll
    for (int q = 0; q < 4; ++q) {
      int ks = q << 5;
      union { uint4 qq[2]; v16h h; } ua, ub;
      ua.qq[0] = *(const uint4*)&Ab[(arow << 7) + ks + koff];
      ua.qq[1] = *(const uint4*)&Ab[(arow << 7) + ks + koff + 8];
      ub.h = bbase[((((s << 7) + ks)) >> 4) + kbh];
      c = __builtin_amdgcn_wmma_f32_16x16x32_f16(
          /*neg_a=*/false, ua.h, /*neg_b=*/false, ub.h,
          /*c_mod=*/(short)0, c, /*reuse_a=*/false, /*reuse_b=*/false);
    }
  }

  // C layout: VGPR r -> M = r (+8 for upper half-wave); N = lane&15 within tile
  int mb = (lane >> 4) << 3;
  #pragma unroll
  for (int r = 0; r < 8; ++r) {
    int M = mb + r;
    float val = c[r] * rsinv[M];
    size_t o = ((size_t)(b * T_) + i0 + M) * F_ + n;
    alignedWS[o] = val;
    outFeat[o] = __logf(val + EPS_);
  }
}

// ---------------- per-row dot/norm partials for window loss ----------------
__global__ void k_rowdots(const float* __restrict__ x,
                          const float* __restrict__ alignedWS,
                          float* __restrict__ dvec,
                          float* __restrict__ f2,
                          float* __restrict__ a2) {
  __shared__ float rd[128], rf[128], ra[128];
  int t = blockIdx.x, b = blockIdx.y, f = threadIdx.x;
  size_t o = ((size_t)(b * T_) + t) * F_ + f;
  float fe = __expf(x[o]);
  float al = alignedWS[o];
  rd[f] = fe * al; rf[f] = fe * fe; ra[f] = al * al;
  __syncthreads();
  for (int o2 = 64; o2 > 0; o2 >>= 1) {
    if (f < o2) { rd[f] += rd[f + o2]; rf[f] += rf[f + o2]; ra[f] += ra[f + o2]; }
    __syncthreads();
  }
  if (f == 0) {
    dvec[b * T_ + t] = rd[0];
    f2[b * T_ + t]   = rf[0];
    a2[b * T_ + t]   = ra[0];
  }
}

// ---------------- sliding-window cosine loss ----------------
__global__ void k_loss(const float* __restrict__ dvec,
                       const float* __restrict__ f2,
                       const float* __restrict__ a2,
                       float* __restrict__ out) {
  __shared__ float red[256];
  int tid = threadIdx.x;
  float acc = 0.f;
  for (int w = tid; w < B_ * NW_; w += 256) {
    int b = w / NW_, s = w - b * NW_;
    float Sd = 0.f, Sf = 0.f, Sa = 0.f;
    #pragma unroll
    for (int i = 0; i < W_; ++i) {
      int t = b * T_ + s + i;
      Sd += dvec[t]; Sf += f2[t]; Sa += a2[t];
    }
    float csim = Sd / (fmaxf(sqrtf(Sf), EPS_) * fmaxf(sqrtf(Sa), EPS_));
    acc += 1.0f - csim;
  }
  red[tid] = acc;
  __syncthreads();
  for (int o = 128; o > 0; o >>= 1) {
    if (tid < o) red[tid] += red[tid + o];
    __syncthreads();
  }
  if (tid == 0) out[0] = red[0] / (float)(B_ * NW_);
}

extern "C" void kernel_launch(void* const* d_in, const int* in_sizes, int n_in,
                              void* d_out, int out_size, void* d_ws, size_t ws_size,
                              hipStream_t stream) {
  (void)in_sizes; (void)n_in; (void)out_size; (void)ws_size;
  const float* x      = (const float*)d_in[0];   // (B,T,F)
  const float* conv_w = (const float*)d_in[1];   // (1,F,5)
  const float* conv_b = (const float*)d_in[2];   // (1,)
  const float* proj   = (const float*)d_in[3];   // (B,T,T)

  float* out      = (float*)d_out;
  float* outFeat  = out;                                   // B*T*F
  float* outScore = out + (size_t)B_ * T_ * F_;            // B*T
  float* outLoss  = outScore + (size_t)B_ * T_;            // 1

  char* ws = (char*)d_ws;
  size_t off = 0;
  auto alloc = [&](size_t bytes) -> void* {
    void* p = ws + off;
    off += (bytes + 255) & ~(size_t)255;
    return p;
  };
  float*    score_raw = (float*)alloc((size_t)B_ * T_ * 4);
  float*    sum0      = (float*)alloc(B_ * 4);
  float*    max0      = (float*)alloc(B_ * 4);
  float*    rowmax    = (float*)alloc((size_t)B_ * T_ * 4);
  float*    rowsum    = (float*)alloc((size_t)B_ * T_ * 4);
  _Float16* featT     = (_Float16*)alloc((size_t)B_ * F_ * T_ * 2);
  float*    alignedWS = (float*)alloc((size_t)B_ * T_ * F_ * 4);
  float*    dvec      = (float*)alloc((size_t)B_ * T_ * 4);
  float*    f2        = (float*)alloc((size_t)B_ * T_ * 4);
  float*    a2        = (float*)alloc((size_t)B_ * T_ * 4);

  k_conv_sigmoid<<<dim3(B_ * T_ / 256), 256, 0, stream>>>(x, conv_w, conv_b, score_raw);
  k_batch_reduce<<<dim3(B_), 256, 0, stream>>>(score_raw, sum0, max0);
  k_score_norm <<<dim3(B_), 256, 0, stream>>>(score_raw, sum0, max0, outScore);
  k_featT      <<<dim3(T_ / 32, F_ / 32, B_), 256, 0, stream>>>(x, featT);
  k_rowstats   <<<dim3(T_, B_), 256, 0, stream>>>(proj, outScore, rowmax, rowsum);
  k_gemm       <<<dim3(T_ / 16, B_), 256, 0, stream>>>(proj, featT, outScore, rowmax,
                                                       rowsum, alignedWS, outFeat);
  k_rowdots    <<<dim3(T_, B_), 128, 0, stream>>>(x, alignedWS, dvec, f2, a2);
  k_loss       <<<1, 256, 0, stream>>>(dvec, f2, a2, outLoss);
}
